// RWKV_Tmix_x070_10849087389839
// MI455X (gfx1250) — matrix operand order repaired and checked by reading, not compiled, not run on hardware
//
#include <hip/hip_runtime.h>
#include <hip/hip_bf16.h>
#include <math.h>

#define B_ 4
#define T_ 4096
#define C_ 1024
#define H_ 16
#define N_ 64
#define M_ (B_ * T_)          // 16384 token rows
#define EPSGN 6.4e-4f         // 1e-5 * 8^2

typedef __attribute__((ext_vector_type(16))) __bf16 bf16x16;
typedef __attribute__((ext_vector_type(8)))  float  f32x8;

enum { EPI_NONE = 0, EPI_TANH, EPI_SIG, EPI_SIG_BIAS, EPI_DECAY, EPI_MV };

__device__ __forceinline__ float sigmoid_(float x) { return 1.0f / (1.0f + expf(-x)); }

// ---------------------------------------------------------------------------
// One-time weight convert: out[n*K + k] (bf16, (N,K) row-major) from
//   trans=0: in (N,K) fp32 row-major (Wr/Wk/Wv/Wo)
//   trans=1: in (K,N) fp32 row-major (low-rank w1/w2/a1/a2/v1/v2/g1/g2)
// ---------------------------------------------------------------------------
__global__ __launch_bounds__(256) void wcvt_kernel(
    const float* __restrict__ in, __bf16* __restrict__ out,
    int K, int N, int trans)
{
  const int id = blockIdx.x * 256 + threadIdx.x;
  if (id >= N * K) return;
  const int n = id / K;
  const int k = id - n * K;
  const float v = trans ? in[(size_t)k * N + n] : in[id];
  out[id] = (__bf16)v;
}

// ---------------------------------------------------------------------------
// Fused GEMM:  Cout[M x N] = epi( mix(A) @ Wb^T )
//   A: M x K fp32 (lda). coef != nullptr => token-shift mix fused on load:
//     a[m,c] = x[m,c] + (x[m-1,c] - x[m,c]) * coef[c]   (prev=0 at t==0)
//   Wb: N x K bf16 row-major (pre-converted).
// Block: 128 threads (4 waves). Block tile 128x64, wave tile 32x64,
// K-step 32, bf16 WMMA with fp32 accumulation.
// ---------------------------------------------------------------------------
template <int EPI>
__global__ __launch_bounds__(128) void wmma_gemm_kernel(
    const float* __restrict__ A, int lda,
    const float* __restrict__ coef,
    const __bf16* __restrict__ Wb,
    const float* __restrict__ bias,   // per-output-col bias (w0/a0/v0) or null
    const float* __restrict__ aux0,   // EPI_MV: v_raw
    const float* __restrict__ aux1,   // EPI_MV: v_first
    float* __restrict__ Cout, int ldc,
    int N, int K)
{
  __shared__ __align__(16) __bf16 ldsA[128 * 40];
  __shared__ __align__(16) __bf16 ldsB[64 * 40];

  const int tid  = threadIdx.x;
  const int wave = tid >> 5;
  const int lane = tid & 31;
  const int mBase = blockIdx.y * 128;
  const int nBase = blockIdx.x * 64;

  f32x8 acc[2][4];
#pragma unroll
  for (int ms = 0; ms < 2; ++ms)
#pragma unroll
    for (int ns = 0; ns < 4; ++ns)
#pragma unroll
      for (int e = 0; e < 8; ++e) acc[ms][ns][e] = 0.0f;

  for (int k0 = 0; k0 < K; k0 += 32) {
    // ---- stage A tile (128 rows x 32 k): global fp32 -> shift-mix -> bf16 LDS
#pragma unroll
    for (int pass = 0; pass < 4; ++pass) {
      const int row = pass * 32 + (tid >> 2);
      const int cb  = (tid & 3) * 8;
      const int gRow = mBase + row;
      const int gc   = k0 + cb;
      const float* ap = A + (size_t)gRow * lda + gc;
      if (k0 + 64 < K) __builtin_prefetch(ap + 64, 0, 0);  // global_prefetch_b8, 2 K-tiles ahead
      float4 a0 = *(const float4*)(ap);
      float4 a1 = *(const float4*)(ap + 4);
      float vals[8] = {a0.x, a0.y, a0.z, a0.w, a1.x, a1.y, a1.z, a1.w};
      if (coef != nullptr) {
        float pv[8];
        const int t = gRow & (T_ - 1);
        if (t > 0) {
          const float* pp = ap - lda;
          float4 p0 = *(const float4*)(pp);
          float4 p1 = *(const float4*)(pp + 4);
          pv[0] = p0.x; pv[1] = p0.y; pv[2] = p0.z; pv[3] = p0.w;
          pv[4] = p1.x; pv[5] = p1.y; pv[6] = p1.z; pv[7] = p1.w;
        } else {
#pragma unroll
          for (int i = 0; i < 8; ++i) pv[i] = 0.0f;
        }
#pragma unroll
        for (int i = 0; i < 8; ++i)
          vals[i] = vals[i] + (pv[i] - vals[i]) * coef[gc + i];
      }
#pragma unroll
      for (int i = 0; i < 8; ++i) ldsA[row * 40 + cb + i] = (__bf16)vals[i];
    }

    // ---- stage B tile (64 n x 32 k): straight bf16 copy, zero-fill n >= N
    {
      const int row = tid >> 1;          // 0..63
      const int kb  = (tid & 1) * 16;    // 0 or 16
      const int gN  = nBase + row;
      uint4 u0 = {0, 0, 0, 0}, u1 = {0, 0, 0, 0};
      if (gN < N) {
        const __bf16* wp = Wb + (size_t)gN * K + k0 + kb;
        u0 = *(const uint4*)(wp);
        u1 = *(const uint4*)(wp + 8);
      }
      *(uint4*)(&ldsB[row * 40 + kb])     = u0;
      *(uint4*)(&ldsB[row * 40 + kb + 8]) = u1;
    }
    __syncthreads();

    // ---- A fragments (16x32): lanes 0-15: M=lane, K {0..7,16..23}; lanes 16-31: K {8..15,24..31}
    union Frag { bf16x16 v; uint4 u[2]; };
    Frag fa[2];
#pragma unroll
    for (int ms = 0; ms < 2; ++ms) {
      const int r   = wave * 32 + ms * 16 + (lane & 15);
      const int klo = (lane >> 4) * 8;
      fa[ms].u[0] = *(const uint4*)(&ldsA[r * 40 + klo]);
      fa[ms].u[1] = *(const uint4*)(&ldsA[r * 40 + 16 + klo]);
    }
#pragma unroll
    for (int ns = 0; ns < 4; ++ns) {
      // B fragment (32x16): lanes 0-15 hold K {0..7,16..23} of column lane; lanes 16-31 K {8..15,24..31}
      Frag fb;
      const int n   = ns * 16 + (lane & 15);
      const int kof = (lane >> 4) * 8;
      fb.u[0] = *(const uint4*)(&ldsB[n * 40 + kof]);
      fb.u[1] = *(const uint4*)(&ldsB[n * 40 + 16 + kof]);
#pragma unroll
      for (int ms = 0; ms < 2; ++ms) {
        acc[ms][ns] = __builtin_amdgcn_wmma_f32_16x16x32_bf16(
            false, fa[ms].v, false, fb.v, (short)0, acc[ms][ns], false, false);
      }
    }
    __syncthreads();
  }

  // ---- epilogue + store. C/D layout: vgpr r, lane<16 -> M=r, lane>=16 -> M=r+8
  const int rowOff = (lane >> 4) * 8;
  const int colIn  = lane & 15;
#pragma unroll
  for (int ms = 0; ms < 2; ++ms) {
#pragma unroll
    for (int ns = 0; ns < 4; ++ns) {
      const int col = nBase + ns * 16 + colIn;
      if (col >= N) continue;
#pragma unroll
      for (int r = 0; r < 8; ++r) {
        const int rowg = mBase + wave * 32 + ms * 16 + rowOff + r;
        const size_t idx = (size_t)rowg * ldc + col;
        float y = acc[ms][ns][r];
        float outv;
        if constexpr (EPI == EPI_NONE) {
          outv = y;
        } else if constexpr (EPI == EPI_TANH) {
          outv = tanhf(y);
        } else if constexpr (EPI == EPI_SIG) {
          outv = sigmoid_(y);
        } else if constexpr (EPI == EPI_SIG_BIAS) {
          outv = sigmoid_(bias[col] + y);
        } else if constexpr (EPI == EPI_DECAY) {
          const float u = bias[col] + y;               // w0 + tanh(..)@w2
          const float w = -log1pf(expf(-u)) - 0.5f;    // -softplus(-u) - 0.5
          outv = expf(-expf(w));                       // per-channel decay
        } else {  // EPI_MV: v = v_raw + (v_first - v_raw) * sigmoid(v0 + y)
          const float s  = sigmoid_(bias[col] + y);
          const float vr = aux0[idx];
          const float vf = aux1[idx];
          outv = vr + (vf - vr) * s;
        }
        Cout[idx] = outv;
      }
    }
  }
}

// ---------------------------------------------------------------------------
// prep: kk = normalize_head(k_raw * k_k); k = k_raw * (1 + (a-1)*k_a)
// one block per token row; 256 threads x 4 channels; head = 16 lanes (wave32)
// ---------------------------------------------------------------------------
__global__ __launch_bounds__(256) void prep_kernel(
    float* __restrict__ kbuf, const float* __restrict__ abuf,
    const float* __restrict__ k_k, const float* __restrict__ k_a,
    float* __restrict__ kkbuf)
{
  const int row = blockIdx.x;
  const int tid = threadIdx.x;
  const int c = tid * 4;
  const size_t base = (size_t)row * C_ + c;

  float4 kr  = *(const float4*)(kbuf + base);
  float4 kkc = *(const float4*)(k_k + c);
  float kk0 = kr.x * kkc.x, kk1 = kr.y * kkc.y, kk2 = kr.z * kkc.z, kk3 = kr.w * kkc.w;
  float ss = kk0 * kk0 + kk1 * kk1 + kk2 * kk2 + kk3 * kk3;
#pragma unroll
  for (int m = 1; m < 16; m <<= 1) ss += __shfl_xor(ss, m, 32);
  const float inv = 1.0f / fmaxf(sqrtf(ss), 1e-12f);

  float4 a4  = *(const float4*)(abuf + base);
  float4 ka4 = *(const float4*)(k_a + c);
  float4 kko = {kk0 * inv, kk1 * inv, kk2 * inv, kk3 * inv};
  float4 kn  = {kr.x * (1.0f + (a4.x - 1.0f) * ka4.x),
                kr.y * (1.0f + (a4.y - 1.0f) * ka4.y),
                kr.z * (1.0f + (a4.z - 1.0f) * ka4.z),
                kr.w * (1.0f + (a4.w - 1.0f) * ka4.w)};
  *(float4*)(kkbuf + base) = kko;
  *(float4*)(kbuf + base)  = kn;
}

// ---------------------------------------------------------------------------
// WKV bidirectional scan. grid = (B*H, 2 dirs), block = 64 threads.
// Thread i owns state row i (64 floats in VGPRs). Per-step vectors via LDS.
//   sa_i = -sum_j S[i,j]*kk[j]
//   S[i,j] = S[i,j]*dec[j] + sa_i*(kk[j]*a[j]) + v[i]*k[j]
//   y[i]   = sum_j S[i,j]*r[j]
// ---------------------------------------------------------------------------
__global__ __launch_bounds__(64) void wkv_scan_kernel(
    const float* __restrict__ r, const float* __restrict__ dec,
    const float* __restrict__ k, const float* __restrict__ v,
    const float* __restrict__ kk, const float* __restrict__ a,
    float* __restrict__ y_f, float* __restrict__ y_b)
{
  const int bh  = blockIdx.x;
  const int dir = blockIdx.y;
  const int b = bh / H_, h = bh % H_;
  const int i = threadIdx.x;

  __shared__ float lr[64], ld_[64], lk[64], lkk[64], lb[64];
  float s[64];
#pragma unroll
  for (int j = 0; j < 64; ++j) s[j] = 0.0f;

  float* yout = dir ? y_b : y_f;
  const int t0 = dir ? (T_ - 1) : 0;
  const int dt = dir ? -1 : 1;

  for (int step = 0; step < T_; ++step) {
    const int t = t0 + dt * step;
    const size_t base = ((size_t)(b * T_ + t)) * C_ + h * N_;
    lr[i]  = r[base + i];
    ld_[i] = dec[base + i];
    lk[i]  = k[base + i];
    const float kkv = kk[base + i];
    lkk[i] = kkv;
    lb[i]  = kkv * a[base + i];
    __syncthreads();

    float sa = 0.0f;
#pragma unroll
    for (int j = 0; j < 64; ++j) sa = fmaf(s[j], lkk[j], sa);
    sa = -sa;

    const float vi = v[base + i];
    float out = 0.0f;
#pragma unroll
    for (int j = 0; j < 64; ++j) {
      float sj = s[j] * ld_[j];
      sj = fmaf(sa, lb[j], sj);
      sj = fmaf(vi, lk[j], sj);
      s[j] = sj;
      out = fmaf(sj, lr[j], out);
    }
    yout[base + i] = out;
    __syncthreads();
  }
}

// ---------------------------------------------------------------------------
// post: blend fwd/bwd, GroupNorm(H), + (r*k*r_k).sum * v, * g  -> z (into yf)
// ---------------------------------------------------------------------------
__global__ __launch_bounds__(256) void post_kernel(
    const float* __restrict__ y_f, const float* __restrict__ y_b,
    const float* __restrict__ r, const float* __restrict__ k,
    const float* __restrict__ v, const float* __restrict__ g,
    const float* __restrict__ r_k, const float* __restrict__ ln_w,
    const float* __restrict__ ln_b, const float* __restrict__ alpha_p,
    float* __restrict__ zout)
{
  const int row = blockIdx.x;
  const int tid = threadIdx.x;
  const int c = tid * 4;
  const int h = c >> 6;
  const int cin = c & 63;
  const size_t base = (size_t)row * C_ + c;
  const float al = *alpha_p;

  float4 yf4 = *(const float4*)(y_f + base);
  float4 yb4 = *(const float4*)(y_b + base);
  float y0 = al * yf4.x + (1.0f - al) * yb4.x;
  float y1 = al * yf4.y + (1.0f - al) * yb4.y;
  float y2 = al * yf4.z + (1.0f - al) * yb4.z;
  float y3 = al * yf4.w + (1.0f - al) * yb4.w;

  float4 r4 = *(const float4*)(r + base);
  float4 k4 = *(const float4*)(k + base);
  float4 v4 = *(const float4*)(v + base);
  float4 g4 = *(const float4*)(g + base);
  float4 rk4 = *(const float4*)(r_k + h * 64 + cin);

  float sum = y0 + y1 + y2 + y3;
  float sq  = y0 * y0 + y1 * y1 + y2 * y2 + y3 * y3;
  float dot = r4.x * k4.x * rk4.x + r4.y * k4.y * rk4.y +
              r4.z * k4.z * rk4.z + r4.w * k4.w * rk4.w;
#pragma unroll
  for (int m = 1; m < 16; m <<= 1) {
    sum += __shfl_xor(sum, m, 32);
    sq  += __shfl_xor(sq, m, 32);
    dot += __shfl_xor(dot, m, 32);
  }
  const float mean = sum * (1.0f / 64.0f);
  const float var  = sq * (1.0f / 64.0f) - mean * mean;
  const float inv  = rsqrtf(var + EPSGN);

  float4 lw = *(const float4*)(ln_w + c);
  float4 lb = *(const float4*)(ln_b + c);
  float4 z;
  z.x = (((y0 - mean) * inv) * lw.x + lb.x + dot * v4.x) * g4.x;
  z.y = (((y1 - mean) * inv) * lw.y + lb.y + dot * v4.y) * g4.y;
  z.z = (((y2 - mean) * inv) * lw.z + lb.z + dot * v4.z) * g4.z;
  z.w = (((y3 - mean) * inv) * lw.w + lb.w + dot * v4.w) * g4.w;
  *(float4*)(zout + base) = z;
}

// ---------------------------------------------------------------------------
extern "C" void kernel_launch(void* const* d_in, const int* in_sizes, int n_in,
                              void* d_out, int out_size, void* d_ws, size_t ws_size,
                              hipStream_t stream)
{
  const float* x       = (const float*)d_in[0];
  const float* v_first = (const float*)d_in[1];
  const float* x_r = (const float*)d_in[2];
  const float* x_w = (const float*)d_in[3];
  const float* x_k = (const float*)d_in[4];
  const float* x_v = (const float*)d_in[5];
  const float* x_a = (const float*)d_in[6];
  const float* x_g = (const float*)d_in[7];
  const float* w0  = (const float*)d_in[8];
  const float* w1  = (const float*)d_in[9];
  const float* w2  = (const float*)d_in[10];
  const float* a0  = (const float*)d_in[11];
  const float* a1  = (const float*)d_in[12];
  const float* a2  = (const float*)d_in[13];
  const float* v0  = (const float*)d_in[14];
  const float* v1  = (const float*)d_in[15];
  const float* v2  = (const float*)d_in[16];
  const float* g1  = (const float*)d_in[17];
  const float* g2  = (const float*)d_in[18];
  const float* k_k = (const float*)d_in[19];
  const float* k_a = (const float*)d_in[20];
  const float* r_k = (const float*)d_in[21];
  const float* Wr  = (const float*)d_in[22];
  const float* Wk  = (const float*)d_in[23];
  const float* Wv  = (const float*)d_in[24];
  const float* Wo  = (const float*)d_in[25];
  const float* lnw = (const float*)d_in[26];
  const float* lnb = (const float*)d_in[27];
  const float* alp = (const float*)d_in[28];

  float* ws = (float*)d_ws;
  const size_t MC = (size_t)M_ * C_;
  float* rb  = ws + 0 * MC;
  float* kb  = ws + 1 * MC;
  float* vb  = ws + 2 * MC;
  float* db  = ws + 3 * MC;   // decay
  float* ab  = ws + 4 * MC;   // sigmoid a
  float* gb  = ws + 5 * MC;   // gate
  float* kkb = ws + 6 * MC;
  float* yf  = ws + 7 * MC;   // fwd scan out, later z
  float* yb  = ws + 8 * MC;   // bwd scan out
  float* t1  = ws + 9 * MC;   // low-rank stage-1 (M x <=160)

  // bf16 weight pool after t1 (t1 byte size is 16B-aligned)
  __bf16* bw = (__bf16*)(ws + 9 * MC + (size_t)M_ * 160);
  __bf16* bWr = bw;                 // 1024x1024
  __bf16* bWk = bWr + 1048576;
  __bf16* bWv = bWk + 1048576;
  __bf16* bWo = bWv + 1048576;
  __bf16* bw1 = bWo + 1048576;      // ( 64,1024)
  __bf16* bw2 = bw1 + 65536;        // (1024,  64)
  __bf16* ba1 = bw2 + 65536;        // ( 64,1024)
  __bf16* ba2 = ba1 + 65536;        // (1024,  64)
  __bf16* bv1 = ba2 + 65536;        // ( 32,1024)
  __bf16* bv2 = bv1 + 32768;        // (1024,  32)
  __bf16* bg1 = bv2 + 32768;        // (160,1024)
  __bf16* bg2 = bg1 + 163840;       // (1024, 160)

  // ---- pre-convert weights to bf16, all in (N,K) row-major
  const int cvtB = 256;
  const int nBig = (1048576 + cvtB - 1) / cvtB;
  wcvt_kernel<<<nBig, cvtB, 0, stream>>>(Wr, bWr, C_, C_, 0);
  wcvt_kernel<<<nBig, cvtB, 0, stream>>>(Wk, bWk, C_, C_, 0);
  wcvt_kernel<<<nBig, cvtB, 0, stream>>>(Wv, bWv, C_, C_, 0);
  wcvt_kernel<<<nBig, cvtB, 0, stream>>>(Wo, bWo, C_, C_, 0);
  wcvt_kernel<<<(65536 + cvtB - 1) / cvtB, cvtB, 0, stream>>>(w1, bw1, C_, 64, 1);
  wcvt_kernel<<<(65536 + cvtB - 1) / cvtB, cvtB, 0, stream>>>(w2, bw2, 64, C_, 1);
  wcvt_kernel<<<(65536 + cvtB - 1) / cvtB, cvtB, 0, stream>>>(a1, ba1, C_, 64, 1);
  wcvt_kernel<<<(65536 + cvtB - 1) / cvtB, cvtB, 0, stream>>>(a2, ba2, 64, C_, 1);
  wcvt_kernel<<<(32768 + cvtB - 1) / cvtB, cvtB, 0, stream>>>(v1, bv1, C_, 32, 1);
  wcvt_kernel<<<(32768 + cvtB - 1) / cvtB, cvtB, 0, stream>>>(v2, bv2, 32, C_, 1);
  wcvt_kernel<<<(163840 + cvtB - 1) / cvtB, cvtB, 0, stream>>>(g1, bg1, C_, 160, 1);
  wcvt_kernel<<<(163840 + cvtB - 1) / cvtB, cvtB, 0, stream>>>(g2, bg2, 160, C_, 1);

  const dim3 blk(128);
  const dim3 gBig(C_ / 64, M_ / 128);  // 16 x 128
  const dim3 g64(1, M_ / 128);
  const dim3 g160(3, M_ / 128);

  // r, k_raw, v_raw : fused token-shift mix + bf16 WMMA GEMM (x @ W.T)
  wmma_gemm_kernel<EPI_NONE><<<gBig, blk, 0, stream>>>(
      x, C_, x_r, bWr, nullptr, nullptr, nullptr, rb, C_, C_, C_);
  wmma_gemm_kernel<EPI_NONE><<<gBig, blk, 0, stream>>>(
      x, C_, x_k, bWk, nullptr, nullptr, nullptr, kb, C_, C_, C_);
  wmma_gemm_kernel<EPI_NONE><<<gBig, blk, 0, stream>>>(
      x, C_, x_v, bWv, nullptr, nullptr, nullptr, vb, C_, C_, C_);

  // decay: tanh(xw @ w1) -> t1 ; dec = exp(-exp(-softplus(-(w0 + t1@w2)) - 0.5))
  wmma_gemm_kernel<EPI_TANH><<<g64, blk, 0, stream>>>(
      x, C_, x_w, bw1, nullptr, nullptr, nullptr, t1, 64, 64, C_);
  wmma_gemm_kernel<EPI_DECAY><<<gBig, blk, 0, stream>>>(
      t1, 64, nullptr, bw2, w0, nullptr, nullptr, db, C_, C_, 64);

  // a = sigmoid(a0 + (xa @ a1) @ a2)
  wmma_gemm_kernel<EPI_NONE><<<g64, blk, 0, stream>>>(
      x, C_, x_a, ba1, nullptr, nullptr, nullptr, t1, 64, 64, C_);
  wmma_gemm_kernel<EPI_SIG_BIAS><<<gBig, blk, 0, stream>>>(
      t1, 64, nullptr, ba2, a0, nullptr, nullptr, ab, C_, C_, 64);

  // v = v_raw + (v_first - v_raw) * sigmoid(v0 + (xv @ v1) @ v2)
  wmma_gemm_kernel<EPI_NONE><<<g64, blk, 0, stream>>>(
      x, C_, x_v, bv1, nullptr, nullptr, nullptr, t1, 32, 32, C_);
  wmma_gemm_kernel<EPI_MV><<<gBig, blk, 0, stream>>>(
      t1, 32, nullptr, bv2, v0, vb, v_first, vb, C_, C_, 32);

  // g = sigmoid(xg @ g1) @ g2
  wmma_gemm_kernel<EPI_SIG><<<g160, blk, 0, stream>>>(
      x, C_, x_g, bg1, nullptr, nullptr, nullptr, t1, 160, 160, C_);
  wmma_gemm_kernel<EPI_NONE><<<gBig, blk, 0, stream>>>(
      t1, 160, nullptr, bg2, nullptr, nullptr, nullptr, gb, C_, C_, 160);

  // kk normalize + k modulation
  prep_kernel<<<M_, 256, 0, stream>>>(kb, ab, k_k, k_a, kkb);

  // bidirectional WKV scan
  wkv_scan_kernel<<<dim3(B_ * H_, 2), 64, 0, stream>>>(rb, db, kb, vb, kkb, ab, yf, yb);

  // blend + GroupNorm + residual + gate -> z (in yf)
  post_kernel<<<M_, 256, 0, stream>>>(yf, yb, rb, kb, vb, gb, r_k, lnw, lnb, alp, yf);

  // out = z @ Wo.T
  wmma_gemm_kernel<EPI_NONE><<<gBig, blk, 0, stream>>>(
      yf, C_, nullptr, bWo, nullptr, nullptr, nullptr, (float*)d_out, C_, C_, C_);

  // second tuple output: v_first passthrough
  hipMemcpyAsync((float*)d_out + MC, v_first, MC * sizeof(float),
                 hipMemcpyDeviceToDevice, stream);
}
